// FaberConv_62723702391592
// MI455X (gfx1250) — matrix operand order repaired and checked by reading, not compiled, not run on hardware
//
#include <hip/hip_runtime.h>

#define NN 100000
#define NE 1600000
#define DIM 64
#define ALPHA_C 0.5f

typedef __attribute__((ext_vector_type(16))) __bf16 v16bf;
typedef __attribute__((ext_vector_type(8)))  float  v8f;

// ---------------------------------------------------------------- atomics
__device__ __forceinline__ void atomAddF(float* p, float v) {
    (void)__hip_atomic_fetch_add(p, v, __ATOMIC_RELAXED, __HIP_MEMORY_SCOPE_AGENT);
}

// ---------------------------------------------------------------- degrees
__global__ void k_degree(const int* __restrict__ row, const int* __restrict__ col,
                         float* __restrict__ dO, float* __restrict__ dI, int E) {
    int e = blockIdx.x * blockDim.x + threadIdx.x;
    if (e < E) {
        atomAddF(&dO[row[e]], 1.0f);
        atomAddF(&dI[col[e]], 1.0f);
    }
}

__global__ void k_degpow(float* __restrict__ dO, float* __restrict__ dI, int N) {
    int i = blockIdx.x * blockDim.x + threadIdx.x;
    if (i < 2 * N) {
        float* p = (i < N) ? dO : dI;
        int j = (i < N) ? i : i - N;
        float d = p[j];
        p[j] = (d > 0.0f) ? rsqrtf(sqrtf(d)) : 0.0f;   // d^-0.25, 0-deg -> 0
    }
}

__global__ void k_edgew(const int* __restrict__ row, const int* __restrict__ col,
                        const float* __restrict__ dpO, const float* __restrict__ dpI,
                        float* __restrict__ w, int E) {
    int e = blockIdx.x * blockDim.x + threadIdx.x;
    if (e < E) w[e] = dpO[row[e]] * dpI[col[e]];
}

// ---------------------------------------------------------------- SpMM: vout[dst] += w * vin[src]
// 16 lanes per edge, float4 per lane, hardware f32 atomics (L2-resident data).
__global__ void k_spmm(const int* __restrict__ dst, const int* __restrict__ src,
                       const float* __restrict__ w, const float* __restrict__ vin,
                       float* __restrict__ vout, int E) {
    int tid = blockIdx.x * blockDim.x + threadIdx.x;
    int e = tid >> 4;
    if (e >= E) return;
    int q = tid & 15;
    int s = src[e], d = dst[e];
    float we = w[e];
    float4 v = ((const float4*)vin)[(size_t)s * 16 + q];
    float* o = vout + (size_t)d * DIM + q * 4;
    atomAddF(o + 0, we * v.x);
    atomAddF(o + 1, we * v.y);
    atomAddF(o + 2, we * v.z);
    atomAddF(o + 3, we * v.w);
}

// ---------------------------------------------------------------- weight / bias prep
// Wc[t][o][k] = coeff_t * W_{sd|ds}[i][o][k] in bf16, t=0..2 sd, t=3..5 ds
__global__ void k_prepw(const float* __restrict__ Wsd, const float* __restrict__ Wds,
                        __bf16* __restrict__ Wc) {
    int idx = blockIdx.x * blockDim.x + threadIdx.x;
    if (idx >= 6 * 4096) return;
    int t = idx >> 12, rem = idx & 4095;
    int i = (t < 3) ? t : (t - 3);
    float scale = (i == 0) ? 1.0f : ((i == 1) ? 0.5f : 0.25f);
    float coeff = ((t < 3) ? ALPHA_C : (1.0f - ALPHA_C)) * scale;
    const float* src = (t < 3) ? Wsd : Wds;
    Wc[idx] = (__bf16)(coeff * src[i * 4096 + rem]);
}

__global__ void k_prepb(const float* __restrict__ bsd, const float* __restrict__ bds,
                        float* __restrict__ btot) {
    int o = threadIdx.x;
    if (o < DIM) {
        float s = 0.0f;
        const float sc[3] = {1.0f, 0.5f, 0.25f};
        #pragma unroll
        for (int i = 0; i < 3; ++i)
            s += ALPHA_C * sc[i] * bsd[i * DIM + o] + (1.0f - ALPHA_C) * sc[i] * bds[i * DIM + o];
        btot[o] = s;
    }
}

__global__ void k_initout(float* __restrict__ out, const float* __restrict__ btot, int total) {
    int i = blockIdx.x * blockDim.x + threadIdx.x;
    if (i < total) out[i] = btot[i & (DIM - 1)];
}

// ---------------------------------------------------------------- WMMA GEMM accumulate
// out(N x 64) += bf16(V1) @ W1^T + bf16(V2) @ W2^T   (coeffs pre-folded into W)
__device__ __forceinline__ v16bf pack16(float4 a, float4 b, float4 c, float4 d) {
    v16bf r;
    r[0] = (__bf16)a.x;  r[1] = (__bf16)a.y;  r[2] = (__bf16)a.z;  r[3] = (__bf16)a.w;
    r[4] = (__bf16)b.x;  r[5] = (__bf16)b.y;  r[6] = (__bf16)b.z;  r[7] = (__bf16)b.w;
    r[8] = (__bf16)c.x;  r[9] = (__bf16)c.y;  r[10] = (__bf16)c.z; r[11] = (__bf16)c.w;
    r[12] = (__bf16)d.x; r[13] = (__bf16)d.y; r[14] = (__bf16)d.z; r[15] = (__bf16)d.w;
    return r;
}

__global__ void k_gemm_acc(const float* __restrict__ V1, const float* __restrict__ V2,
                           const __bf16* __restrict__ W1, const __bf16* __restrict__ W2,
                           float* __restrict__ out, int N) {
    const int lane = threadIdx.x & 31;             // wave32
    const int wave = threadIdx.x >> 5;
    const int tile = blockIdx.x * 4 + wave;        // 16 rows per wave
    const int rowbase = tile * 16;
    if (rowbase >= N) return;                      // N % 16 == 0 -> full tiles only
    const int m    = lane & 15;                    // A: row, B/C/D: column
    const int half = lane >> 4;

    v8f acc[4] = {v8f{0.f}, v8f{0.f}, v8f{0.f}, v8f{0.f}};

    const float4* r1 = (const float4*)(V1 + (size_t)(rowbase + m) * DIM);
    const float4* r2 = (const float4*)(V2 + (size_t)(rowbase + m) * DIM);

    #pragma unroll
    for (int kc = 0; kc < 2; ++kc) {               // K = 64 = 2 x 32
        // A-frag 16-bit layout: elements j -> K = kc*32 + half*16 + j
        int q = kc * 8 + half * 4;
        v16bf A1 = pack16(r1[q], r1[q + 1], r1[q + 2], r1[q + 3]);
        v16bf A2 = pack16(r2[q], r2[q + 1], r2[q + 2], r2[q + 3]);
        #pragma unroll
        for (int ot = 0; ot < 4; ++ot) {           // 64 outputs = 4 x 16-col tiles
            // B-frag: lane column o = ot*16+m, elements j -> K = kc*32 + half*16 + j
            v16bf B1 = ((const v16bf*)(W1 + (size_t)(ot * 16 + m) * DIM))[kc * 2 + half];
            acc[ot] = __builtin_amdgcn_wmma_f32_16x16x32_bf16(
                false, A1, false, B1, (short)0, acc[ot], false, false);
            v16bf B2 = ((const v16bf*)(W2 + (size_t)(ot * 16 + m) * DIM))[kc * 2 + half];
            acc[ot] = __builtin_amdgcn_wmma_f32_16x16x32_bf16(
                false, A2, false, B2, (short)0, acc[ot], false, false);
        }
    }

    // C/D layout: element v -> row = half*8 + v, column = ot*16 + m
    #pragma unroll
    for (int ot = 0; ot < 4; ++ot) {
        #pragma unroll
        for (int v = 0; v < 8; ++v) {
            size_t idx = (size_t)(rowbase + half * 8 + v) * DIM + ot * 16 + m;
            out[idx] += acc[ot][v];
        }
    }
}

// ---------------------------------------------------------------- host
extern "C" void kernel_launch(void* const* d_in, const int* in_sizes, int n_in,
                              void* d_out, int out_size, void* d_ws, size_t ws_size,
                              hipStream_t stream) {
    const float* x   = (const float*)d_in[0];
    const int*   ei  = (const int*)d_in[1];
    const float* Wsd = (const float*)d_in[2];
    const float* bsd = (const float*)d_in[3];
    const float* Wds = (const float*)d_in[4];
    const float* bds = (const float*)d_in[5];
    float* out = (float*)d_out;
    const int* row = ei;            // edge_index[0]
    const int* col = ei + NE;       // edge_index[1]

    char* ws = (char*)d_ws;
    size_t off = 0;
    auto alloc = [&](size_t bytes) -> void* {
        void* p = ws + off;
        off = (off + bytes + 255) & ~(size_t)255;
        return p;
    };
    const size_t featBytes = (size_t)NN * DIM * sizeof(float);
    float*  w_edge = (float*)alloc((size_t)NE * sizeof(float));
    float*  dpO    = (float*)alloc((size_t)NN * sizeof(float));
    float*  dpI    = (float*)alloc((size_t)NN * sizeof(float));
    float*  btot   = (float*)alloc(DIM * sizeof(float));
    __bf16* Wc     = (__bf16*)alloc((size_t)6 * 4096 * sizeof(__bf16));
    float*  Y0     = (float*)alloc(featBytes);
    float*  Y1     = (float*)alloc(featBytes);
    float*  Yt     = (float*)alloc(featBytes);

    // --- normalization weights
    hipMemsetAsync(dpO, 0, (size_t)NN * sizeof(float), stream);
    hipMemsetAsync(dpI, 0, (size_t)NN * sizeof(float), stream);
    k_degree<<<(NE + 255) / 256, 256, 0, stream>>>(row, col, dpO, dpI, NE);
    k_degpow<<<(2 * NN + 255) / 256, 256, 0, stream>>>(dpO, dpI, NN);
    k_edgew<<<(NE + 255) / 256, 256, 0, stream>>>(row, col, dpO, dpI, w_edge, NE);

    // --- fold stage coefficients into bf16 weights / single bias vector
    k_prepw<<<(6 * 4096 + 255) / 256, 256, 0, stream>>>(Wsd, Wds, Wc);
    k_prepb<<<1, 64, 0, stream>>>(bsd, bds, btot);
    k_initout<<<(NN * DIM + 255) / 256, 256, 0, stream>>>(out, btot, NN * DIM);

    const int spmmGrid = ((NE * 16) + 255) / 256;
    const int gemmGrid = ((NN / 16) + 3) / 4;      // 4 wave-tiles per 128-thread block

    // --- stage 0: y0 = A x ; yt0 = A^T x
    hipMemsetAsync(Y0, 0, featBytes, stream);
    k_spmm<<<spmmGrid, 256, 0, stream>>>(row, col, w_edge, x, Y0, NE);
    hipMemsetAsync(Yt, 0, featBytes, stream);
    k_spmm<<<spmmGrid, 256, 0, stream>>>(col, row, w_edge, x, Yt, NE);
    k_gemm_acc<<<gemmGrid, 128, 0, stream>>>(Y0, Yt, Wc + 0 * 4096, Wc + 3 * 4096, out, NN);

    // --- stage 1: y1 = A y0 ; yt1 = A^T y1 (reference propagates fresh y)
    hipMemsetAsync(Y1, 0, featBytes, stream);
    k_spmm<<<spmmGrid, 256, 0, stream>>>(row, col, w_edge, Y0, Y1, NE);
    hipMemsetAsync(Yt, 0, featBytes, stream);
    k_spmm<<<spmmGrid, 256, 0, stream>>>(col, row, w_edge, Y1, Yt, NE);
    k_gemm_acc<<<gemmGrid, 128, 0, stream>>>(Y1, Yt, Wc + 1 * 4096, Wc + 4 * 4096, out, NN);

    // --- stage 2: y2 = A y1 ; yt2 = A^T y2
    hipMemsetAsync(Y0, 0, featBytes, stream);
    k_spmm<<<spmmGrid, 256, 0, stream>>>(row, col, w_edge, Y1, Y0, NE);
    hipMemsetAsync(Yt, 0, featBytes, stream);
    k_spmm<<<spmmGrid, 256, 0, stream>>>(col, row, w_edge, Y0, Yt, NE);
    k_gemm_acc<<<gemmGrid, 128, 0, stream>>>(Y0, Yt, Wc + 2 * 4096, Wc + 5 * 4096, out, NN);
}
